// GMSolver_56495999811911
// MI455X (gfx1250) — matrix operand order repaired and checked by reading, not compiled, not run on hardware
//
#include <hip/hip_runtime.h>
#include <hip/hip_bf16.h>
#include <cstdint>
#include <cstddef>

// ---------------------------------------------------------------------------
// Problem constants (reference): N=48 nodes, E=N*(N-1)=2256 edges, H=32, EPS=20
// Heavy math: five eeadj (E x E) GEMMs -> bf16 WMMA with f32 accumulate.
// GEMM dims padded so the WMMA loop is branch-free:
//   M: 2256 -> 2304 (36 x 64-row wave strips), K: 2256 -> 2272 (71 x 32)
// B operand stored transposed (F-major) so fragments load as b128.
// ---------------------------------------------------------------------------
constexpr int  KN   = 48;
constexpr int  KE   = 2256;
constexpr int  GM   = 2304;   // padded M (rows of eeadj / output)
constexpr int  GK   = 2272;   // padded K (reduction dim)
constexpr float EPSV = 20.0f;

// --- output offsets (floats, flat concat in tuple return order) ------------
constexpr size_t O_BI    = 0;                       // (48,1,48)
constexpr size_t O_BIJ   = O_BI    + 2304;          // (2256,1,48,48)
constexpr size_t O_MSGI  = O_BIJ   + (size_t)KE * 2304;
constexpr size_t O_MSGJ  = O_MSGI  + (size_t)KE * 48;
constexpr size_t O_V     = O_MSGJ  + (size_t)KE * 48;
constexpr size_t O_CDUAL = O_V     + 48;
constexpr size_t O_NMEM  = O_CDUAL + 1;             // (48,64,48)
constexpr size_t O_EMEM  = O_NMEM  + (size_t)48 * 64 * 48;
constexpr size_t O_F1    = O_EMEM  + (size_t)KE * 64 * 48;
constexpr size_t O_F2    = O_F1    + 2304;

// ---------------------------------------------------------------------------
// bf16 helpers + WMMA fragment types
// ---------------------------------------------------------------------------
typedef __attribute__((ext_vector_type(16))) __bf16        v16bf;
typedef __attribute__((ext_vector_type(8)))  float         v8f;
typedef __attribute__((ext_vector_type(4)))  unsigned int  u32x4;

union FragAB { v16bf v; u32x4 q[2]; unsigned int w[8]; unsigned short h[16]; };

__device__ __forceinline__ unsigned short f2bf(float x) {
  unsigned int u = __float_as_uint(x);
  u += 0x7fffu + ((u >> 16) & 1u);          // round-to-nearest-even
  return (unsigned short)(u >> 16);
}

// ---------------------------------------------------------------------------
// Core GEMM: C[GM,F] = A[GM,GK](bf16) x Bt[F,GK](bf16, transposed), f32 acc.
// One wave computes a 64x16 strip: 4 M-tiles share one B fragment.
// Fragment layouts per CDNA5 ISA (CDNA5 ISA):
//   A 16x32: lane(l) row = l%16, elem t -> k = (t/8)*16 + (l/16)*8 + t%8
//            -> two contiguous 16B runs at k = half*8 and 16+half*8
//   B 32x16: lane(l) col = l%16, elem t -> k = (t/8)*16 + (l/16)*8 + t%8
//            -> two contiguous 16B runs in the transposed layout (same as A)
//   C/D    : elem r -> row = r + 8*(l/16), col = l%16
// Inner loop: 10 x b128 loads, 2 pointer bumps, 4 WMMAs. No branches.
// ---------------------------------------------------------------------------
__global__ void __launch_bounds__(128)
wmma_gemm_bf16(const unsigned short* __restrict__ A,    // GM x GK
               const unsigned short* __restrict__ Bt,   // F x GK (transposed)
               float* __restrict__ C,                   // GM x F
               int F) {
  const int lane   = threadIdx.x & 31;
  const int wave   = threadIdx.x >> 5;
  const int tilesF = F >> 4;
  const int wid    = blockIdx.x * 4 + wave;      // over (GM/64) x tilesF
  const int mG     = wid / tilesF;               // 64-row strip index
  const int fT     = wid - mG * tilesF;
  const int half   = lane >> 4;
  const int lr     = lane & 15;

  const unsigned short* pa = A  + (size_t)(mG * 64 + lr) * GK + half * 8;
  const unsigned short* pb = Bt + (size_t)(fT * 16 + lr) * GK + half * 8;

  v8f acc0 = {}, acc1 = {}, acc2 = {}, acc3 = {};
  for (int k0 = 0; k0 < GK; k0 += 32) {
    FragAB b, a0, a1, a2, a3;
    b.q[0]  = *(const u32x4*)(pb);
    b.q[1]  = *(const u32x4*)(pb + 16);
    a0.q[0] = *(const u32x4*)(pa);
    a0.q[1] = *(const u32x4*)(pa + 16);
    a1.q[0] = *(const u32x4*)(pa + (size_t)16 * GK);
    a1.q[1] = *(const u32x4*)(pa + (size_t)16 * GK + 16);
    a2.q[0] = *(const u32x4*)(pa + (size_t)32 * GK);
    a2.q[1] = *(const u32x4*)(pa + (size_t)32 * GK + 16);
    a3.q[0] = *(const u32x4*)(pa + (size_t)48 * GK);
    a3.q[1] = *(const u32x4*)(pa + (size_t)48 * GK + 16);
    pa += 32; pb += 32;
    if (k0 + 32 < GK) __builtin_prefetch(pa, 0, 1);     // global_prefetch_b8
    acc0 = __builtin_amdgcn_wmma_f32_16x16x32_bf16(false, a0.v, false, b.v, (short)0, acc0, false, false);
    acc1 = __builtin_amdgcn_wmma_f32_16x16x32_bf16(false, a1.v, false, b.v, (short)0, acc1, false, false);
    acc2 = __builtin_amdgcn_wmma_f32_16x16x32_bf16(false, a2.v, false, b.v, (short)0, acc2, false, false);
    acc3 = __builtin_amdgcn_wmma_f32_16x16x32_bf16(false, a3.v, false, b.v, (short)0, acc3, false, false);
  }
  float* Cb = C + (size_t)(mG * 64) * F + fT * 16 + lr;
#pragma unroll
  for (int r = 0; r < 8; ++r) {
    const size_t rr = (size_t)(r + half * 8) * F;
    Cb[rr]                  = acc0[r];
    Cb[rr + (size_t)16 * F] = acc1[r];
    Cb[rr + (size_t)32 * F] = acc2[r];
    Cb[rr + (size_t)48 * F] = acc3[r];
  }
}

// Zero the 16 padded K entries of every column of the transposed operand.
__global__ void zero_pad_k(unsigned short* __restrict__ Bt, int F) {
  int idx = blockIdx.x * blockDim.x + threadIdx.x;
  if (idx >= F * (GK - KE)) return;
  int col = idx / (GK - KE), k = idx % (GK - KE);
  Bt[(size_t)col * GK + KE + k] = 0;
}

// eeadj (E x E, f32) -> padded bf16 (GM x GK), zeros outside
__global__ void adj_to_bf16_k(const float* __restrict__ in, unsigned short* __restrict__ out) {
  int idx = blockIdx.x * blockDim.x + threadIdx.x;
  const int total = GM * GK;
  if (idx >= total) return;
  int row = idx / GK, col = idx % GK;
  out[idx] = (row < KE && col < KE) ? f2bf(in[(size_t)row * KE + col]) : (unsigned short)0;
}

// ---------------------------------------------------------------------------
// Setup: derive src/dst per edge (argmax of one-hot rows), deterministic
// per-node edge lists (no float atomics -> deterministic output).
// ---------------------------------------------------------------------------
__global__ void setup_idx_k(const float* __restrict__ n2e0,
                            const float* __restrict__ n2e1,
                            int* __restrict__ src, int* __restrict__ dst) {
  int e = blockIdx.x * blockDim.x + threadIdx.x;
  if (e >= KE) return;
  int s = 0, d = 0; float bs = -1.f, bd = -1.f;
  for (int n = 0; n < KN; ++n) {
    float a = n2e0[e * KN + n]; if (a > bs) { bs = a; s = n; }
    float b = n2e1[e * KN + n]; if (b > bd) { bd = b; d = n; }
  }
  src[e] = s; dst[e] = d;
}

__global__ void build_lists_k(const int* __restrict__ src, const int* __restrict__ dst,
                              int* __restrict__ bySrc, int* __restrict__ byDst,
                              int* __restrict__ cs, int* __restrict__ cd) {
  int n = threadIdx.x;
  if (n >= KN) return;
  int a = 0, b = 0;
  for (int e = 0; e < KE; ++e) {
    if (src[e] == n && a < 64) bySrc[n * 64 + a++] = e;
    if (dst[e] == n && b < 64) byDst[n * 64 + b++] = e;
  }
  cs[n] = a; cd[n] = b;
}

// ---------------------------------------------------------------------------
// cmsgi/cmsgj: stable logsumexp over bij rows/cols (+ bi gathered by dst/src)
// One block per edge; bij tile staged in LDS.
// ---------------------------------------------------------------------------
__global__ void __launch_bounds__(128)
cmsg_k(const float* __restrict__ bi, const float* __restrict__ bij,
       const int* __restrict__ src, const int* __restrict__ dst,
       float* __restrict__ cmsgi, float* __restrict__ cmsgj) {
  __shared__ float tile[2304];
  __shared__ float bs[48], bd[48];
  const int e = blockIdx.x;
  const float* be = bij + (size_t)e * 2304;
  for (int i = threadIdx.x; i < 2304; i += 128) tile[i] = be[i];
  if (threadIdx.x < 48) {
    bs[threadIdx.x] = bi[src[e] * 48 + threadIdx.x];
    bd[threadIdx.x] = bi[dst[e] * 48 + threadIdx.x];
  }
  __syncthreads();
  const int t = threadIdx.x;
  if (t < 48) {                      // cmsgi[e, i=t] = lse_j(bd[j] + bij[i,j])
    float m = -1e30f;
    for (int j = 0; j < 48; ++j) m = fmaxf(m, bd[j] + tile[t * 48 + j]);
    float s = 0.f;
    for (int j = 0; j < 48; ++j) s += __expf((bd[j] + tile[t * 48 + j] - m) * EPSV);
    cmsgi[(size_t)e * 48 + t] = m + __logf(s) * (1.f / EPSV);
  } else if (t >= 64 && t < 112) {   // cmsgj[e, j] = lse_i(bs[i] + bij[i,j])
    int j = t - 64;
    float m = -1e30f;
    for (int i = 0; i < 48; ++i) m = fmaxf(m, bs[i] + tile[i * 48 + j]);
    float s = 0.f;
    for (int i = 0; i < 48; ++i) s += __expf((bs[i] + tile[i * 48 + j] - m) * EPSV);
    cmsgj[(size_t)e * 48 + j] = m + __logf(s) * (1.f / EPSV);
  }
}

// ---------------------------------------------------------------------------
// Feature builders: write the transposed bf16 operand Bt[col*GK + e].
// Thread mapping is col-major so Bt writes are contiguous along e.
// ---------------------------------------------------------------------------
__global__ void edge_feature_k(const float* __restrict__ msgi, const float* __restrict__ msgj,
                               const float* __restrict__ cmsgi, const float* __restrict__ cmsgj,
                               const float* __restrict__ bi,
                               const int* __restrict__ src, const int* __restrict__ dst,
                               unsigned short* __restrict__ Bt) {
  int idx = blockIdx.x * blockDim.x + threadIdx.x;
  const int total = 6 * 48 * KE;
  if (idx >= total) return;
  int e = idx % KE, col = idx / KE;
  int p = col % 48, c = col / 48;
  float v;
  switch (c) {
    case 0: v = msgi[e * 48 + p];  break;
    case 1: v = msgj[e * 48 + p];  break;
    case 2: v = cmsgi[e * 48 + p]; break;
    case 3: v = cmsgj[e * 48 + p]; break;
    case 4: v = bi[src[e] * 48 + p]; break;
    default: v = bi[dst[e] * 48 + p]; break;
  }
  Bt[(size_t)col * GK + e] = f2bf(v);
}

// elstm input: [efeat(32) | nfeat[src](32) | nfeat[dst](32) | emem0.h(32)]
__global__ void edge_lstm_input_k(const float* __restrict__ efeat,
                                  const float* __restrict__ nfeat,
                                  const float* __restrict__ emem0,
                                  const int* __restrict__ src, const int* __restrict__ dst,
                                  unsigned short* __restrict__ Bt) {
  int idx = blockIdx.x * blockDim.x + threadIdx.x;
  const int total = 128 * 48 * KE;
  if (idx >= total) return;
  int e = idx % KE, col = idx / KE;
  int p = col % 48, c = col / 48;
  float v;
  if (c < 32)       v = efeat[((size_t)e * 32 + c) * 48 + p];
  else if (c < 64)  v = nfeat[((size_t)src[e] * 32 + (c - 32)) * 48 + p];
  else if (c < 96)  v = nfeat[((size_t)dst[e] * 32 + (c - 64)) * 48 + p];
  else              v = emem0[((size_t)e * 64 + 32 + (c - 96)) * 48 + p];
  Bt[(size_t)col * GK + e] = f2bf(v);
}

// epp input: [emem_new.h(32) | nmem_new.h[src](32) | nmem_new.h[dst](32)]
__global__ void edge_pp_input_k(const float* __restrict__ emem,
                                const float* __restrict__ nmem,
                                const int* __restrict__ src, const int* __restrict__ dst,
                                unsigned short* __restrict__ Bt) {
  int idx = blockIdx.x * blockDim.x + threadIdx.x;
  const int total = 96 * 48 * KE;
  if (idx >= total) return;
  int e = idx % KE, col = idx / KE;
  int p = col % 48, c = col / 48;
  float v;
  if (c < 32)       v = emem[((size_t)e * 64 + 32 + c) * 48 + p];
  else if (c < 64)  v = nmem[((size_t)src[e] * 64 + 32 + (c - 32)) * 48 + p];
  else              v = nmem[((size_t)dst[e] * 64 + 32 + (c - 64)) * 48 + p];
  Bt[(size_t)col * GK + e] = f2bf(v);
}

// ---------------------------------------------------------------------------
// Edge conv1d (k=3, SAME) over last axis; input = Y (post-map), LDS-staged.
// outb (optional) goes to the transposed bf16 operand for the next GEMM.
// ---------------------------------------------------------------------------
__global__ void __launch_bounds__(192)
edge_conv_k(const float* __restrict__ Y, const float* __restrict__ W,
            const float* __restrict__ bias, int Cin, int Cout, int relu,
            unsigned short* __restrict__ outb, float* __restrict__ outf) {
  extern __shared__ float sx[];                 // Cin * 50, zero-padded ends
  const int e = blockIdx.x;
  const float* ye = Y + (size_t)e * Cin * 48;
  for (int i = threadIdx.x; i < Cin; i += 192) { sx[i * 50] = 0.f; sx[i * 50 + 49] = 0.f; }
  for (int i = threadIdx.x; i < Cin * 48; i += 192) sx[(i / 48) * 50 + 1 + (i % 48)] = ye[i];
  __syncthreads();
  const int p = threadIdx.x % 48;
  for (int o = threadIdx.x / 48; o < Cout; o += 4) {
    float acc = bias[o];
    const float* w = W + (size_t)o * Cin * 3;
    for (int i = 0; i < Cin; ++i) {
      const float* sp = sx + i * 50 + p;
      acc = fmaf(w[i * 3 + 0], sp[0], acc);
      acc = fmaf(w[i * 3 + 1], sp[1], acc);
      acc = fmaf(w[i * 3 + 2], sp[2], acc);
    }
    if (relu) acc = fmaxf(acc, 0.f);
    if (outb) outb[(size_t)(o * 48 + p) * GK + e] = f2bf(acc);
    if (outf) outf[((size_t)e * Cout + o) * 48 + p] = acc;
  }
}

// ---------------------------------------------------------------------------
// Fused conv(128->128,k3) + LSTM cell (edges: grid=E, nodes: grid=48)
// ---------------------------------------------------------------------------
__global__ void __launch_bounds__(384)
lstm_conv_k(const float* __restrict__ XM, const float* __restrict__ W,
            const float* __restrict__ bias, const float* __restrict__ mem0,
            float* __restrict__ memOut) {
  __shared__ float sx[128 * 50];
  const int r = blockIdx.x;
  const float* xr = XM + (size_t)r * 128 * 48;
  for (int i = threadIdx.x; i < 128; i += 384) { sx[i * 50] = 0.f; sx[i * 50 + 49] = 0.f; }
  for (int i = threadIdx.x; i < 128 * 48; i += 384) sx[(i / 48) * 50 + 1 + (i % 48)] = xr[i];
  __syncthreads();
  const int p = threadIdx.x % 48;
  for (int c = threadIdx.x / 48; c < 32; c += 8) {
    float g[4];
#pragma unroll
    for (int q = 0; q < 4; ++q) {                   // gates: i, f, g, o
      const int oc = q * 32 + c;
      float acc = bias[oc];
      const float* w = W + (size_t)oc * 128 * 3;
      for (int i = 0; i < 128; ++i) {
        const float* sp = sx + i * 50 + p;
        acc = fmaf(w[i * 3 + 0], sp[0], acc);
        acc = fmaf(w[i * 3 + 1], sp[1], acc);
        acc = fmaf(w[i * 3 + 2], sp[2], acc);
      }
      g[q] = acc;
    }
    const float cold = mem0[((size_t)r * 64 + c) * 48 + p];
    const float ig = 1.f / (1.f + __expf(-g[0]));
    const float fg = 1.f / (1.f + __expf(-g[1]));
    const float gg = tanhf(g[2]);
    const float og = 1.f / (1.f + __expf(-g[3]));
    const float c2 = fg * cold + ig * gg;
    const float h2 = og * tanhf(c2);
    memOut[((size_t)r * 64 + c) * 48 + p]      = c2;
    memOut[((size_t)r * 64 + c + 32) * 48 + p] = h2;
  }
}

// ---------------------------------------------------------------------------
// Node-side small kernels (48 rows -> plain VALU, negligible FLOPs)
// ---------------------------------------------------------------------------
__global__ void node_map_k(const float* __restrict__ adj, const float* __restrict__ X,
                           float* __restrict__ out, int C) {
  int idx = blockIdx.x * blockDim.x + threadIdx.x;
  int total = 48 * C * 48;
  if (idx >= total) return;
  int p = idx % 48, c = (idx / 48) % C, n = idx / (48 * C);
  float acc = 0.f;
  const float* ar = adj + n * 48;
  for (int m = 0; m < 48; ++m) acc = fmaf(ar[m], X[((size_t)m * C + c) * 48 + p], acc);
  out[idx] = acc;
}

__global__ void node_conv_k(const float* __restrict__ XM, const float* __restrict__ W,
                            const float* __restrict__ bias, float* __restrict__ out,
                            int Cin, int Cout, int relu) {
  int idx = blockIdx.x * blockDim.x + threadIdx.x;
  int total = 48 * Cout * 48;
  if (idx >= total) return;
  int p = idx % 48, o = (idx / 48) % Cout, n = idx / (48 * Cout);
  float acc = bias[o];
  const float* xb = XM + (size_t)n * Cin * 48;
  for (int i = 0; i < Cin; ++i) {
    const float* w = W + ((size_t)o * Cin + i) * 3;
    if (p > 0)  acc = fmaf(w[0], xb[i * 48 + p - 1], acc);
    acc = fmaf(w[1], xb[i * 48 + p], acc);
    if (p < 47) acc = fmaf(w[2], xb[i * 48 + p + 1], acc);
  }
  if (relu) acc = fmaxf(acc, 0.f);
  out[idx] = acc;
}

// node_feature = [bi, bi+v, nnmsgi, nnmsgj, ncmsgi, ncmsgj, f1, f2]; also f1/f2 out
__global__ void node_features_k(const float* __restrict__ bi, const float* __restrict__ v,
                                const float* __restrict__ bij,
                                const float* __restrict__ msgi, const float* __restrict__ msgj,
                                const float* __restrict__ cmsgi, const float* __restrict__ cmsgj,
                                const int* __restrict__ deco,
                                const int* __restrict__ bySrc, const int* __restrict__ byDst,
                                const int* __restrict__ cs, const int* __restrict__ cd,
                                float* __restrict__ nf, float* __restrict__ f1o,
                                float* __restrict__ f2o) {
  int n = blockIdx.x, p = threadIdx.x;
  if (p >= 48) return;
  float smi = 0, smj = 0, sci = 0, scj = 0, sf1 = 0, sf2 = 0;
  int dn = deco[n];
  int a = cs[n];
  for (int k = 0; k < a; ++k) {
    int e = bySrc[n * 64 + k];
    smi += msgi[(size_t)e * 48 + p];
    sci += cmsgi[(size_t)e * 48 + p];
    sf1 += bij[(size_t)e * 2304 + dn * 48 + p];
  }
  int b = cd[n];
  for (int k = 0; k < b; ++k) {
    int e = byDst[n * 64 + k];
    smj += msgj[(size_t)e * 48 + p];
    scj += cmsgj[(size_t)e * 48 + p];
    sf2 += bij[(size_t)e * 2304 + p * 48 + dn];
  }
  float bv = bi[n * 48 + p];
  float* o = nf + (size_t)n * 8 * 48;
  o[0 * 48 + p] = bv;        o[1 * 48 + p] = bv + v[p];
  o[2 * 48 + p] = smi;       o[3 * 48 + p] = smj;
  o[4 * 48 + p] = sci;       o[5 * 48 + p] = scj;
  o[6 * 48 + p] = sf1;       o[7 * 48 + p] = sf2;
  f1o[n * 48 + p] = sf1;     f2o[n * 48 + p] = sf2;
}

// nlstm input: [nfeat(32) | sum_src efeat(32) | sum_dst efeat(32) | nmem0.h(32)]
__global__ void node_lstm_input_k(const float* __restrict__ nfeat, const float* __restrict__ efeat,
                                  const float* __restrict__ nmem0,
                                  const int* __restrict__ bySrc, const int* __restrict__ byDst,
                                  const int* __restrict__ cs, const int* __restrict__ cd,
                                  float* __restrict__ out) {
  int n = blockIdx.x;
  for (int idx = threadIdx.x; idx < 128 * 48; idx += blockDim.x) {
    int c = idx / 48, p = idx % 48;
    float v;
    if (c < 32) v = nfeat[((size_t)n * 32 + c) * 48 + p];
    else if (c < 64) {
      int cc = c - 32; float s = 0; int a = cs[n];
      for (int k = 0; k < a; ++k) s += efeat[((size_t)bySrc[n * 64 + k] * 32 + cc) * 48 + p];
      v = s;
    } else if (c < 96) {
      int cc = c - 64; float s = 0; int a = cd[n];
      for (int k = 0; k < a; ++k) s += efeat[((size_t)byDst[n * 64 + k] * 32 + cc) * 48 + p];
      v = s;
    } else v = nmem0[((size_t)n * 64 + 32 + (c - 96)) * 48 + p];
    out[((size_t)n * 128 + c) * 48 + p] = v;
  }
}

// pp input: [nmem_new.h(32) | sum_src emem_new.h(32) | sum_dst emem_new.h(32)]
__global__ void node_pp_input_k(const float* __restrict__ nmem, const float* __restrict__ emem,
                                const int* __restrict__ bySrc, const int* __restrict__ byDst,
                                const int* __restrict__ cs, const int* __restrict__ cd,
                                float* __restrict__ out) {
  int n = blockIdx.x;
  for (int idx = threadIdx.x; idx < 96 * 48; idx += blockDim.x) {
    int c = idx / 48, p = idx % 48;
    float v;
    if (c < 32) v = nmem[((size_t)n * 64 + 32 + c) * 48 + p];
    else if (c < 64) {
      int cc = c - 32; float s = 0; int a = cs[n];
      for (int k = 0; k < a; ++k) s += emem[((size_t)bySrc[n * 64 + k] * 64 + 32 + cc) * 48 + p];
      v = s;
    } else {
      int cc = c - 64; float s = 0; int a = cd[n];
      for (int k = 0; k < a; ++k) s += emem[((size_t)byDst[n * 64 + k] * 64 + 32 + cc) * 48 + p];
      v = s;
    }
    out[((size_t)n * 96 + c) * 48 + p] = v;
  }
}

// ---------------------------------------------------------------------------
// Final elementwise + reductions
// ---------------------------------------------------------------------------
__global__ void nv_k(const float* __restrict__ npp, const float* __restrict__ v,
                     float* __restrict__ nv, float* __restrict__ vOut) {
  int p = threadIdx.x;
  if (p >= 48) return;
  float s = 0.f;
  for (int n = 0; n < 48; ++n) s += npp[n * 48 + p];
  s *= (1.f / 48.f);
  nv[p] = s; vOut[p] = v[p] + s;
}

__global__ void msgs_k(const float* __restrict__ ef2, const float* __restrict__ cmsgi,
                       const float* __restrict__ cmsgj, const float* __restrict__ msgi,
                       const float* __restrict__ msgj,
                       float* __restrict__ nmsgi, float* __restrict__ nmsgj,
                       float* __restrict__ msgiO, float* __restrict__ msgjO) {
  int idx = blockIdx.x * blockDim.x + threadIdx.x;
  const int total = KE * 48;
  if (idx >= total) return;
  int e = idx / 48, p = idx % 48;
  float ni = ef2[(size_t)e * 96 + p]      + 0.5f * cmsgi[idx] - msgi[idx];
  float nj = ef2[(size_t)e * 96 + 48 + p] + 0.5f * cmsgj[idx] - msgj[idx];
  nmsgi[idx] = ni;             nmsgj[idx] = nj;
  msgiO[idx] = msgi[idx] + ni; msgjO[idx] = msgj[idx] + nj;
}

__global__ void bij_out_k(const float* __restrict__ bij, const float* __restrict__ nmsgi,
                          const float* __restrict__ nmsgj, float* __restrict__ out) {
  int idx = blockIdx.x * blockDim.x + threadIdx.x;
  const int total = KE * 2304;
  if (idx >= total) return;
  int e = idx / 2304, r = (idx % 2304) / 48, j = idx % 48;
  out[idx] = bij[idx] - nmsgi[(size_t)e * 48 + r] - nmsgj[(size_t)e * 48 + j];
}

__global__ void bi_out_k(const float* __restrict__ bi, const float* __restrict__ nmsgi,
                         const float* __restrict__ nmsgj, const float* __restrict__ nv,
                         const int* __restrict__ bySrc, const int* __restrict__ byDst,
                         const int* __restrict__ cs, const int* __restrict__ cd,
                         float* __restrict__ out) {
  int n = blockIdx.x, p = threadIdx.x;
  if (p >= 48) return;
  float s = bi[n * 48 + p] - nv[p];
  int a = cs[n];
  for (int k = 0; k < a; ++k) s += nmsgi[(size_t)bySrc[n * 64 + k] * 48 + p];
  int b = cd[n];
  for (int k = 0; k < b; ++k) s += nmsgj[(size_t)byDst[n * 64 + k] * 48 + p];
  out[n * 48 + p] = s;
}

__global__ void __launch_bounds__(256)
edge_lse_k(const float* __restrict__ x, float* __restrict__ out) {
  __shared__ float red[256];
  const int e = blockIdx.x;
  const float* xe = x + (size_t)e * 2304;
  float m = -1e30f;
  for (int i = threadIdx.x; i < 2304; i += 256) m = fmaxf(m, xe[i]);
  red[threadIdx.x] = m; __syncthreads();
  for (int s = 128; s > 0; s >>= 1) {
    if (threadIdx.x < s) red[threadIdx.x] = fmaxf(red[threadIdx.x], red[threadIdx.x + s]);
    __syncthreads();
  }
  m = red[0]; __syncthreads();
  float sum = 0.f;
  for (int i = threadIdx.x; i < 2304; i += 256) sum += __expf((xe[i] - m) * EPSV);
  red[threadIdx.x] = sum; __syncthreads();
  for (int s = 128; s > 0; s >>= 1) {
    if (threadIdx.x < s) red[threadIdx.x] += red[threadIdx.x + s];
    __syncthreads();
  }
  if (threadIdx.x == 0) out[e] = m + __logf(red[0]) * (1.f / EPSV);
}

__global__ void node_lse_k(const float* __restrict__ biOut, float* __restrict__ out) {
  int n = threadIdx.x;
  if (n >= 48) return;
  const float* x = biOut + n * 48;
  float m = -1e30f;
  for (int j = 0; j < 48; ++j) m = fmaxf(m, x[j]);
  float s = 0.f;
  for (int j = 0; j < 48; ++j) s += __expf((x[j] - m) * EPSV);
  out[n] = m + __logf(s) * (1.f / EPSV);
}

__global__ void __launch_bounds__(256)
cdual_k(const float* __restrict__ eLse, const float* __restrict__ nLse,
        const float* __restrict__ vOut, float* __restrict__ out) {
  __shared__ float red[256];
  float s = 0.f;
  for (int i = threadIdx.x; i < KE; i += 256) s += eLse[i];
  if (threadIdx.x < 48) s += nLse[threadIdx.x] + vOut[threadIdx.x];
  red[threadIdx.x] = s; __syncthreads();
  for (int st = 128; st > 0; st >>= 1) {
    if (threadIdx.x < st) red[threadIdx.x] += red[threadIdx.x + st];
    __syncthreads();
  }
  if (threadIdx.x == 0) out[0] = red[0];
}

// ---------------------------------------------------------------------------
// Host: orchestrate the whole forward pass on `stream`.
// ---------------------------------------------------------------------------
extern "C" void kernel_launch(void* const* d_in, const int* in_sizes, int n_in,
                              void* d_out, int out_size, void* d_ws, size_t ws_size,
                              hipStream_t stream) {
  (void)in_sizes; (void)n_in; (void)out_size; (void)ws_size;
  const float* bi     = (const float*)d_in[0];
  const float* bij    = (const float*)d_in[1];
  const float* msgi   = (const float*)d_in[2];
  const float* msgj   = (const float*)d_in[3];
  const float* vin    = (const float*)d_in[4];
  const float* nmem0  = (const float*)d_in[5];
  const float* emem0  = (const float*)d_in[6];
  const float* neadj  = (const float*)d_in[7];
  const float* eeadj  = (const float*)d_in[8];
  const float* n2e0   = (const float*)d_in[11];
  const float* n2e1   = (const float*)d_in[12];
  const int*   deco   = (const int*)d_in[13];
  const float* fm_w0  = (const float*)d_in[14]; const float* fm_b0 = (const float*)d_in[15];
  const float* fm_w1  = (const float*)d_in[16]; const float* fm_b1 = (const float*)d_in[17];
  const float* efm_w0 = (const float*)d_in[18]; const float* efm_b0 = (const float*)d_in[19];
  const float* efm_w1 = (const float*)d_in[20]; const float* efm_b1 = (const float*)d_in[21];
  const float* nls_w  = (const float*)d_in[22]; const float* nls_b = (const float*)d_in[23];
  const float* els_w  = (const float*)d_in[24]; const float* els_b = (const float*)d_in[25];
  const float* pp_w0  = (const float*)d_in[26]; const float* pp_b0 = (const float*)d_in[27];
  const float* pp_w1  = (const float*)d_in[28]; const float* pp_b1 = (const float*)d_in[29];
  const float* epp_w0 = (const float*)d_in[30]; const float* epp_b0 = (const float*)d_in[31];
  const float* epp_w1 = (const float*)d_in[32]; const float* epp_b1 = (const float*)d_in[33];

  float* out = (float*)d_out;
  float* dBi   = out + O_BI;    float* dBij  = out + O_BIJ;
  float* dMsgi = out + O_MSGI;  float* dMsgj = out + O_MSGJ;
  float* dV    = out + O_V;     float* dCd   = out + O_CDUAL;
  float* dNmem = out + O_NMEM;  float* dEmem = out + O_EMEM;
  float* dF1   = out + O_F1;    float* dF2   = out + O_F2;

  // --- workspace carve-up (256B aligned) ---
  char* base = (char*)d_ws;
  size_t off = 0;
  auto alloc = [&](size_t bytes) -> char* {
    char* p = base + off;
    off = (off + bytes + 255) & ~(size_t)255;
    return p;
  };
  int*            srcI  = (int*)alloc((size_t)KE * 4);
  int*            dstI  = (int*)alloc((size_t)KE * 4);
  int*            bySrc = (int*)alloc(48 * 64 * 4);
  int*            byDst = (int*)alloc(48 * 64 * 4);
  int*            csN   = (int*)alloc(48 * 4);
  int*            cdN   = (int*)alloc(48 * 4);
  unsigned short* adjB  = (unsigned short*)alloc((size_t)GM * GK * 2);       // padded bf16 eeadj
  unsigned short* XB    = (unsigned short*)alloc((size_t)128 * 48 * GK * 2); // transposed bf16 operand
  float*          Yb    = (float*)alloc((size_t)GM * 128 * 48 * 4);          // padded GEMM output
  float*          efeat = (float*)alloc((size_t)KE * 32 * 48 * 4);
  float*          cmsgi = (float*)alloc((size_t)KE * 48 * 4);
  float*          cmsgj = (float*)alloc((size_t)KE * 48 * 4);
  float*          nmsgi = (float*)alloc((size_t)KE * 48 * 4);
  float*          nmsgj = (float*)alloc((size_t)KE * 48 * 4);
  float*          nodeA = (float*)alloc((size_t)48 * 128 * 48 * 4);
  float*          nodeB = (float*)alloc((size_t)48 * 128 * 48 * 4);
  float*          nf8   = (float*)alloc((size_t)48 * 8 * 48 * 4);
  float*          nfeat = (float*)alloc((size_t)48 * 32 * 48 * 4);
  float*          nvbuf = (float*)alloc(48 * 4);
  float*          eLse  = (float*)alloc((size_t)KE * 4);
  float*          nLse  = (float*)alloc(48 * 4);

  auto cdiv = [](long long a, long long b) { return (int)((a + b - 1) / b); };
  // Zero the 16 padded K entries per column of the operand, then GEMM.
  auto gemm = [&](unsigned short* Bt, float* C, int F) {
    zero_pad_k<<<cdiv((long long)F * (GK - KE), 256), 256, 0, stream>>>(Bt, F);
    wmma_gemm_bf16<<<9 * (F / 16), 128, 0, stream>>>(adjB, Bt, C, F);
  };

  // ---- setup ----
  setup_idx_k<<<cdiv(KE, 256), 256, 0, stream>>>(n2e0, n2e1, srcI, dstI);
  build_lists_k<<<1, 64, 0, stream>>>(srcI, dstI, bySrc, byDst, csN, cdN);
  adj_to_bf16_k<<<cdiv((long long)GM * GK, 256), 256, 0, stream>>>(eeadj, adjB);
  cmsg_k<<<KE, 128, 0, stream>>>(bi, bij, srcI, dstI, cmsgi, cmsgj);

  // ---- edge CNN (efm): two map+conv stages via WMMA GEMMs ----
  edge_feature_k<<<cdiv((long long)KE * 6 * 48, 256), 256, 0, stream>>>(
      msgi, msgj, cmsgi, cmsgj, bi, srcI, dstI, XB);
  gemm(XB, Yb, 6 * 48);
  edge_conv_k<<<KE, 192, 6 * 50 * 4, stream>>>(Yb, efm_w0, efm_b0, 6, 32, 1, XB, nullptr);
  gemm(XB, Yb, 32 * 48);
  edge_conv_k<<<KE, 192, 32 * 50 * 4, stream>>>(Yb, efm_w1, efm_b1, 32, 32, 1, nullptr, efeat);

  // ---- node CNN (fm) ----
  node_features_k<<<48, 48, 0, stream>>>(bi, vin, bij, msgi, msgj, cmsgi, cmsgj,
                                         deco, bySrc, byDst, csN, cdN, nf8, dF1, dF2);
  node_map_k<<<cdiv(48 * 8 * 48, 256), 256, 0, stream>>>(neadj, nf8, nodeA, 8);
  node_conv_k<<<cdiv(48 * 32 * 48, 256), 256, 0, stream>>>(nodeA, fm_w0, fm_b0, nodeB, 8, 32, 1);
  node_map_k<<<cdiv(48 * 32 * 48, 256), 256, 0, stream>>>(neadj, nodeB, nodeA, 32);
  node_conv_k<<<cdiv(48 * 32 * 48, 256), 256, 0, stream>>>(nodeA, fm_w1, fm_b1, nfeat, 32, 32, 1);

  // ---- edge LSTM: [efeat | nfeat[src] | nfeat[dst] | emem0.h] -> gates -> mem
  edge_lstm_input_k<<<cdiv((long long)KE * 128 * 48, 256), 256, 0, stream>>>(
      efeat, nfeat, emem0, srcI, dstI, XB);

  // ---- node LSTM ----
  node_lstm_input_k<<<48, 256, 0, stream>>>(nfeat, efeat, nmem0, bySrc, byDst, csN, cdN, nodeB);
  node_map_k<<<cdiv(48 * 128 * 48, 256), 256, 0, stream>>>(neadj, nodeB, nodeA, 128);
  lstm_conv_k<<<48, 384, 0, stream>>>(nodeA, nls_w, nls_b, nmem0, dNmem);

  gemm(XB, Yb, 128 * 48);
  lstm_conv_k<<<KE, 384, 0, stream>>>(Yb, els_w, els_b, emem0, dEmem);

  // ---- post-processing CNNs (epp / pp) ----
  edge_pp_input_k<<<cdiv((long long)KE * 96 * 48, 256), 256, 0, stream>>>(
      dEmem, dNmem, srcI, dstI, XB);
  gemm(XB, Yb, 96 * 48);
  edge_conv_k<<<KE, 192, 96 * 50 * 4, stream>>>(Yb, epp_w0, epp_b0, 96, 16, 1, XB, nullptr);
  gemm(XB, Yb, 16 * 48);
  edge_conv_k<<<KE, 192, 16 * 50 * 4, stream>>>(Yb, epp_w1, epp_b1, 16, 2, 0, nullptr, efeat);

  node_pp_input_k<<<48, 256, 0, stream>>>(dNmem, dEmem, bySrc, byDst, csN, cdN, nodeB);
  node_map_k<<<cdiv(48 * 96 * 48, 256), 256, 0, stream>>>(neadj, nodeB, nodeA, 96);
  node_conv_k<<<cdiv(48 * 16 * 48, 256), 256, 0, stream>>>(nodeA, pp_w0, pp_b0, nodeB, 96, 16, 1);
  node_map_k<<<cdiv(48 * 16 * 48, 256), 256, 0, stream>>>(neadj, nodeB, nodeA, 16);
  node_conv_k<<<cdiv(48 * 1 * 48, 256), 256, 0, stream>>>(nodeA, pp_w1, pp_b1, nodeB, 16, 1, 0);
  nv_k<<<1, 64, 0, stream>>>(nodeB, vin, nvbuf, dV);

  // ---- dual updates + outputs ----
  msgs_k<<<cdiv((long long)KE * 48, 256), 256, 0, stream>>>(
      efeat, cmsgi, cmsgj, msgi, msgj, nmsgi, nmsgj, dMsgi, dMsgj);
  bij_out_k<<<cdiv((long long)KE * 2304, 256), 256, 0, stream>>>(bij, nmsgi, nmsgj, dBij);
  bi_out_k<<<48, 48, 0, stream>>>(bi, nmsgi, nmsgj, nvbuf, bySrc, byDst, csN, cdN, dBi);

  // ---- cdual ----
  edge_lse_k<<<KE, 256, 0, stream>>>(dBij, eLse);
  node_lse_k<<<1, 64, 0, stream>>>(dBi, nLse);
  cdual_k<<<1, 256, 0, stream>>>(eLse, nLse, dV, dCd);
}
